// GraphEncoder_29643864277456
// MI455X (gfx1250) — matrix operand order repaired and checked by reading, not compiled, not run on hardware
//
#include <hip/hip_runtime.h>
#include <hip/hip_bf16.h>
#include <math.h>

typedef __attribute__((ext_vector_type(16))) _Float16 v16h;
typedef __attribute__((ext_vector_type(8)))  _Float16 v8h;
typedef __attribute__((ext_vector_type(8)))  float    v8f;

#define NGRAPHS 512

// ---------- float atomic-max via order-preserving uint encoding ----------
__device__ __forceinline__ unsigned encF(float f) {
  unsigned u = __float_as_uint(f);
  return (u & 0x80000000u) ? ~u : (u | 0x80000000u);
}
__device__ __forceinline__ float decF(unsigned u) {
  u = (u & 0x80000000u) ? (u & 0x7FFFFFFFu) : ~u;
  return __uint_as_float(u);
}

// ---------- generic zero ----------
__global__ void k_zero(float* p, size_t n) {
  size_t i = (size_t)blockIdx.x * blockDim.x + threadIdx.x;
  if (i < n) p[i] = 0.0f;
}

// ---------- weight prep: Wt[c*K + k] = (f16) W[k*Ncols + c] ----------
__global__ void k_cvt_wt(const float* __restrict__ W, _Float16* __restrict__ Wt,
                         int K, int Ncols) {
  int t = blockIdx.x * blockDim.x + threadIdx.x;
  if (t >= K * Ncols) return;
  int k = t / Ncols, c = t - k * Ncols;
  Wt[(size_t)c * K + k] = (_Float16)W[t];
}

// ---------- activation prep: A16[t] = (f16)(in[t] + bias[t & cmask]) ----------
__global__ void k_cvt_a(const float* __restrict__ in, const float* __restrict__ bias,
                        _Float16* __restrict__ out, long long total, int cmask) {
  long long t = (long long)blockIdx.x * blockDim.x + threadIdx.x;
  if (t >= total) return;
  float v = in[t];
  if (bias) v += bias[(int)(t & cmask)];
  out[t] = (_Float16)v;
}

// ---------- fused embedding gather -> f16: A16[n,c] = (f16) emb[x[n],c] ----------
__global__ void k_embed_f16(const int* __restrict__ x, const float* __restrict__ emb,
                            _Float16* __restrict__ A16, int n) {
  int t = blockIdx.x * blockDim.x + threadIdx.x;
  if (t >= n * 64) return;
  int node = t >> 6, c = t & 63;
  A16[t] = (_Float16)emb[(size_t)x[node] * 64 + c];
}

// ---------- WMMA GEMM (f16 in, f32 acc): out[N,COUT] = A16 @ Wt^T (+out_bias) ----------
// A16: row-major [N,CIN] f16 ; Wt: [COUT,CIN] f16 (i.e. W transposed)
// A fragment: elem j -> K = (j>=8?16:0) + hi*8 + (j&7)  => two b128 loads/lane
// B fragment: col = lane&15, elem j -> K = (j>=8?16:0) + hi*8 + (j&7) => two b128 loads/lane
template<int CIN, int COUT>
__global__ __launch_bounds__(256) void k_gemm16(const _Float16* __restrict__ A16,
                                                const _Float16* __restrict__ Wt,
                                                const float* __restrict__ out_bias,
                                                float* __restrict__ out, int nrows) {
  constexpr int NT = COUT / 16;
  int wave = blockIdx.x * (blockDim.x >> 5) + (threadIdx.x >> 5);
  int lane = threadIdx.x & 31;
  int m0 = wave * 16;
  if (m0 >= nrows) return;                 // wave-uniform exit, EXEC stays all-1 for WMMA
  int row = lane & 15;
  int hi  = lane >> 4;
  int rowIdx = m0 + row;
  if (rowIdx >= nrows) rowIdx = nrows - 1; // clamp tail-row loads
  v8f acc[NT] = {};
#pragma unroll
  for (int kk = 0; kk < CIN; kk += 32) {
    const v8h* ap = (const v8h*)(A16 + (size_t)rowIdx * CIN + kk + hi * 8);
    v8h alo = ap[0];                       // K = kk + hi*8 + 0..7
    v8h ahi = ap[2];                       // K = kk + 16 + hi*8 + 0..7
    v16h a = __builtin_shufflevector(alo, ahi, 0, 1, 2, 3, 4, 5, 6, 7,
                                     8, 9, 10, 11, 12, 13, 14, 15);
#pragma unroll
    for (int t = 0; t < NT; ++t) {
      int col = t * 16 + row;
      const v8h* bp = (const v8h*)(Wt + (size_t)col * CIN + kk + hi * 8);
      v8h blo = bp[0];                     // K = kk + hi*8 + 0..7
      v8h bhi = bp[2];                     // K = kk + 16 + hi*8 + 0..7
      v16h b = __builtin_shufflevector(blo, bhi, 0, 1, 2, 3, 4, 5, 6, 7,
                                       8, 9, 10, 11, 12, 13, 14, 15);
      acc[t] = __builtin_amdgcn_wmma_f32_16x16x32_f16(
          false, a, false, b, (short)0, acc[t], false, false);
    }
  }
#pragma unroll
  for (int t = 0; t < NT; ++t) {
    int col = t * 16 + row;
    float ob = out_bias ? out_bias[col] : 0.0f;
#pragma unroll
    for (int r = 0; r < 8; ++r) {
      int rr = m0 + hi * 8 + r;            // C/D layout: row = hi*8 + vgpr, col = lane&15
      if (rr < nrows) out[(size_t)rr * COUT + col] = acc[t][r] + ob;
    }
  }
}

// ---------- alpha_s[n,h] = sum_c h[n,h,c]*a_src[h,c]; one wave per node ----------
__global__ __launch_bounds__(256) void k_alpha(const float* __restrict__ h,
                                               const float* __restrict__ a_src,
                                               const float* __restrict__ a_dst,
                                               float* __restrict__ as_out,
                                               float* __restrict__ ad_out, int n) {
  int wave = blockIdx.x * (blockDim.x >> 5) + (threadIdx.x >> 5);
  int lane = threadIdx.x & 31;
  if (wave >= n) return;
  int c4 = lane * 4;                 // lanes 0-15 -> head0 channels, 16-31 -> head1
  int head = c4 >> 6;
  int cc = c4 & 63;
  float4 v = *(const float4*)(h + (size_t)wave * 128 + c4);
  const float* asv = a_src + head * 64 + cc;
  const float* adv = a_dst + head * 64 + cc;
  float ps = v.x * asv[0] + v.y * asv[1] + v.z * asv[2] + v.w * asv[3];
  float pd = v.x * adv[0] + v.y * adv[1] + v.z * adv[2] + v.w * adv[3];
#pragma unroll
  for (int m = 1; m <= 8; m <<= 1) {       // reduce within each 16-lane half (one head each)
    ps += __shfl_xor(ps, m, 16);
    pd += __shfl_xor(pd, m, 16);
  }
  if ((lane & 15) == 0) {
    as_out[wave * 2 + head] = ps;
    ad_out[wave * 2 + head] = pd;
  }
}

// ---------- edge pass 1: e = leaky_relu(as[src]+ad[dst]); segment max over dst ----------
__global__ void k_edge_max(const int* __restrict__ ei, int E, int n,
                           const float* __restrict__ as, const float* __restrict__ ad,
                           float* __restrict__ e_buf, unsigned* __restrict__ mEnc) {
  int t = blockIdx.x * blockDim.x + threadIdx.x;
  int ET = E + n;
  if (t >= ET) return;
  int s = (t < E) ? ei[t]     : (t - E);
  int d = (t < E) ? ei[E + t] : (t - E);
#pragma unroll
  for (int h = 0; h < 2; ++h) {
    float e = as[s * 2 + h] + ad[d * 2 + h];
    e = (e > 0.0f) ? e : 0.2f * e;
    e_buf[(size_t)t * 2 + h] = e;
    atomicMax(&mEnc[d * 2 + h], encF(e));
  }
}

// ---------- edge pass 2: ex = exp(e - m[dst]); segment sum; e_buf <- ex ----------
__global__ void k_edge_sum(const int* __restrict__ ei, int E, int n,
                           const unsigned* __restrict__ mEnc,
                           float* __restrict__ e_buf, float* __restrict__ ssum) {
  int t = blockIdx.x * blockDim.x + threadIdx.x;
  int ET = E + n;
  if (t >= ET) return;
  int d = (t < E) ? ei[E + t] : (t - E);
#pragma unroll
  for (int h = 0; h < 2; ++h) {
    float ex = expf(e_buf[(size_t)t * 2 + h] - decF(mEnc[d * 2 + h]));
    e_buf[(size_t)t * 2 + h] = ex;
    atomicAdd(&ssum[d * 2 + h], ex);
  }
}

// ---------- edge pass 3: agg[dst] += h[src]*alpha ; 1 wave per edge, 4 ch/lane ----------
__global__ void k_edge_aggr(const int* __restrict__ ei, int E, int n,
                            const float* __restrict__ h, const float* __restrict__ e_buf,
                            const float* __restrict__ ssum, float* __restrict__ agg) {
  long long tid = (long long)blockIdx.x * blockDim.x + threadIdx.x;
  int t = (int)(tid >> 5);
  int lane = (int)(tid & 31);
  int ET = E + n;
  if (t >= ET) return;
  int s = (t < E) ? ei[t]     : (t - E);
  int d = (t < E) ? ei[E + t] : (t - E);
  int c4 = lane * 4;
  int head = c4 >> 6;
  float w = e_buf[(size_t)t * 2 + head] / (ssum[d * 2 + head] + 1e-16f);
  float4 v = *(const float4*)(h + (size_t)s * 128 + c4);
  float* o = agg + (size_t)d * 128 + c4;
  atomicAdd(o + 0, v.x * w);
  atomicAdd(o + 1, v.y * w);
  atomicAdd(o + 2, v.z * w);
  atomicAdd(o + 3, v.w * w);
}

// ---------- (agg + bias) -> LayerNorm -> GELU(exact); one wave per node ----------
__global__ __launch_bounds__(256) void k_ln_gelu(const float* __restrict__ agg,
                                                 const float* __restrict__ bias,
                                                 const float* __restrict__ g,
                                                 const float* __restrict__ bb,
                                                 float* __restrict__ out, int n) {
  int wave = blockIdx.x * (blockDim.x >> 5) + (threadIdx.x >> 5);
  int lane = threadIdx.x & 31;
  if (wave >= n) return;
  int c4 = lane * 4;
  float4 v = *(const float4*)(agg + (size_t)wave * 128 + c4);
  float x0 = v.x + bias[c4 + 0], x1 = v.y + bias[c4 + 1];
  float x2 = v.z + bias[c4 + 2], x3 = v.w + bias[c4 + 3];
  float sum = x0 + x1 + x2 + x3;
  float sq  = x0 * x0 + x1 * x1 + x2 * x2 + x3 * x3;
#pragma unroll
  for (int m = 1; m <= 16; m <<= 1) {
    sum += __shfl_xor(sum, m, 32);
    sq  += __shfl_xor(sq,  m, 32);
  }
  float mu  = sum * (1.0f / 128.0f);
  float var = sq * (1.0f / 128.0f) - mu * mu;
  float inv = rsqrtf(var + 1e-5f);
  float y[4] = {x0, x1, x2, x3};
  float4 r;
  float* rp = &r.x;
#pragma unroll
  for (int i = 0; i < 4; ++i) {
    float z = (y[i] - mu) * inv * g[c4 + i] + bb[c4 + i];
    rp[i] = 0.5f * z * (1.0f + erff(z * 0.70710678118654752f));
  }
  *(float4*)(out + (size_t)wave * 128 + c4) = r;
}

// ---------- pooling: segment sum + count over batch_idx ----------
__global__ void k_pool_scatter(const float* __restrict__ hout, const int* __restrict__ batch,
                               float* __restrict__ zsum, float* __restrict__ cnt, int n) {
  int t = blockIdx.x * blockDim.x + threadIdx.x;
  if (t >= n * 64) return;
  int node = t >> 6, c = t & 63;
  int b = batch[node];
  atomicAdd(&zsum[(size_t)b * 64 + c], hout[t]);
  if (c == 0) atomicAdd(&cnt[b], 1.0f);
}

__global__ void k_pool_final(const float* __restrict__ zsum, const float* __restrict__ cnt,
                             float* __restrict__ z) {
  int t = blockIdx.x * blockDim.x + threadIdx.x;
  if (t >= NGRAPHS * 64) return;
  z[t] = zsum[t] / fmaxf(cnt[t >> 6], 1.0f);
}

static inline int gdiv(long long a, long long b) { return (int)((a + b - 1) / b); }

extern "C" void kernel_launch(void* const* d_in, const int* in_sizes, int n_in,
                              void* d_out, int out_size, void* d_ws, size_t ws_size,
                              hipStream_t stream) {
  const int*   x      = (const int*)  d_in[0];
  const int*   ei     = (const int*)  d_in[1];
  const int*   batch  = (const int*)  d_in[2];
  const float* emb    = (const float*)d_in[3];
  const float* W1     = (const float*)d_in[4];
  const float* a_src1 = (const float*)d_in[5];
  const float* a_dst1 = (const float*)d_in[6];
  const float* b1     = (const float*)d_in[7];
  const float* ln_g   = (const float*)d_in[8];
  const float* ln_b   = (const float*)d_in[9];
  const float* W2     = (const float*)d_in[10];
  const float* a_src2 = (const float*)d_in[11];
  const float* a_dst2 = (const float*)d_in[12];
  const float* b2     = (const float*)d_in[13];
  const float* Wp     = (const float*)d_in[14];
  const float* bp     = (const float*)d_in[15];

  const int N  = in_sizes[0];
  const int E  = in_sizes[1] / 2;
  const int ET = E + N;

  // workspace layout (float units). Zero-region [oB2 .. oCT+NG) is contiguous.
  float* ws = (float*)d_ws;
  size_t oB1 = 0;                        // h1 / hg (f32, N*128)
  size_t oB2 = oB1 + (size_t)N * 128;    // agg1 / h2
  size_t oB3 = oB2 + (size_t)N * 128;    // agg2
  size_t oAS = oB3 + (size_t)N * 128;
  size_t oAD = oAS + (size_t)N * 2;
  size_t oM  = oAD + (size_t)N * 2;      // encoded segment max (uint)
  size_t oS  = oM  + (size_t)N * 2;      // segment sum
  size_t oZS = oS  + (size_t)N * 2;      // pooled sums
  size_t oCT = oZS + (size_t)NGRAPHS * 64;
  size_t oZend = oCT + NGRAPHS;
  size_t oA16 = oZend;                   // f16 activations, N*128 halves = N*64 floats
  size_t oEB  = oA16 + (size_t)N * 64;   // per-edge e / ex (ET*2)
  size_t oWt1 = oEB + (size_t)ET * 2;    // f16 W1^T: 128*64 halves
  size_t oWt2 = oWt1 + 4096;             // f16 W2^T: 128*128 halves
  size_t oWtp = oWt2 + 8192;             // f16 Wp^T: 64*128 halves

  float* B1 = ws + oB1;
  float* B2 = ws + oB2;
  float* B3 = ws + oB3;
  float* aS = ws + oAS;
  float* aD = ws + oAD;
  unsigned* mE = (unsigned*)(ws + oM);
  float* sS = ws + oS;
  float* zS = ws + oZS;
  float* cT = ws + oCT;
  _Float16* A16 = (_Float16*)(ws + oA16);
  float* eB = ws + oEB;
  _Float16* Wt1 = (_Float16*)(ws + oWt1);
  _Float16* Wt2 = (_Float16*)(ws + oWt2);
  _Float16* Wtp = (_Float16*)(ws + oWtp);

  float* houtD = (float*)d_out;            // h: N x 64
  float* zD    = houtD + (size_t)N * 64;   // z: 512 x 64

  const int BLK = 256;
  int gemmBlocks = gdiv(gdiv(N, 16), 8);   // 8 waves/block, 16 rows/wave
  int waveBlocks = gdiv(N, 8);             // 1 node/wave, 8 waves/block

  // 0) zero agg buffers, max/sum, pooling (contiguous region)
  k_zero<<<gdiv((long long)(oZend - oB2), BLK), BLK, 0, stream>>>(B2, oZend - oB2);

  // 0b) weight prep (f16 transposed)
  k_cvt_wt<<<gdiv(64 * 128, BLK), BLK, 0, stream>>>(W1, Wt1, 64, 128);
  k_cvt_wt<<<gdiv(128 * 128, BLK), BLK, 0, stream>>>(W2, Wt2, 128, 128);
  k_cvt_wt<<<gdiv(128 * 64, BLK), BLK, 0, stream>>>(Wp, Wtp, 128, 64);

  // 1) embedding -> f16 activations
  k_embed_f16<<<gdiv((long long)N * 64, BLK), BLK, 0, stream>>>(x, emb, A16, N);

  // ---- layer 1 ----
  k_gemm16<64, 128><<<gemmBlocks, BLK, 0, stream>>>(A16, Wt1, nullptr, B1, N);
  k_alpha<<<waveBlocks, BLK, 0, stream>>>(B1, a_src1, a_dst1, aS, aD, N);
  k_edge_max<<<gdiv(ET, BLK), BLK, 0, stream>>>(ei, E, N, aS, aD, eB, mE);
  k_edge_sum<<<gdiv(ET, BLK), BLK, 0, stream>>>(ei, E, N, mE, eB, sS);
  k_edge_aggr<<<gdiv((long long)ET * 32, BLK), BLK, 0, stream>>>(ei, E, N, B1, eB, sS, B2);

  // 2) bias + LayerNorm + GELU -> B1 (f32), then f16 copy for GEMM2
  k_ln_gelu<<<waveBlocks, BLK, 0, stream>>>(B2, b1, ln_g, ln_b, B1, N);
  k_cvt_a<<<gdiv((long long)N * 128, BLK), BLK, 0, stream>>>(B1, nullptr, A16, (long long)N * 128, 127);

  // ---- layer 2 ----
  k_gemm16<128, 128><<<gemmBlocks, BLK, 0, stream>>>(A16, Wt2, nullptr, B2, N);
  k_alpha<<<waveBlocks, BLK, 0, stream>>>(B2, a_src2, a_dst2, aS, aD, N);
  // re-zero segment max/sum (contiguous) and agg2
  k_zero<<<gdiv((long long)N * 4, BLK), BLK, 0, stream>>>((float*)mE, (size_t)N * 4);
  k_zero<<<gdiv((long long)N * 128, BLK), BLK, 0, stream>>>(B3, (size_t)N * 128);
  k_edge_max<<<gdiv(ET, BLK), BLK, 0, stream>>>(ei, E, N, aS, aD, eB, mE);
  k_edge_sum<<<gdiv(ET, BLK), BLK, 0, stream>>>(ei, E, N, mE, eB, sS);
  k_edge_aggr<<<gdiv((long long)ET * 32, BLK), BLK, 0, stream>>>(ei, E, N, B2, eB, sS, B3);

  // 3) projection: (agg2 + b2) @ Wp + bp -> d_out h   (b2 fused into f16 convert)
  k_cvt_a<<<gdiv((long long)N * 128, BLK), BLK, 0, stream>>>(B3, b2, A16, (long long)N * 128, 127);
  k_gemm16<128, 64><<<gemmBlocks, BLK, 0, stream>>>(A16, Wtp, bp, houtD, N);

  // 4) graph mean pooling -> d_out z
  k_pool_scatter<<<gdiv((long long)N * 64, BLK), BLK, 0, stream>>>(houtD, batch, zS, cT, N);
  k_pool_final<<<gdiv(NGRAPHS * 64, BLK), BLK, 0, stream>>>(zS, cT, zD);
}
